// AMN_QP_85727547228420
// MI455X (gfx1250) — compile-verified, matrix operand order checked by reading, not run on hardware
//
#include <hip/hip_runtime.h>

typedef __bf16 bf16;
typedef __attribute__((ext_vector_type(16))) __bf16 v16bf;
typedef __attribute__((ext_vector_type(8)))  __bf16 v8bf;
typedef __attribute__((ext_vector_type(8)))  float  v8f;
typedef __attribute__((ext_vector_type(4)))  float  v4f;

#define WMMA_BF16(acc, a, b)                                                   \
  acc = __builtin_amdgcn_wmma_f32_16x16x32_bf16(false, (a), false, (b),        \
                                                (short)0, (acc), false, false)

// ---------------------------------------------------------------------------
// Pack a K x N matrix (optionally transposed view, optionally scaled) into
// WMMA B-operand fragment layout, bf16.
// Tile (kt, nt): 32 lanes x 16 bf16 contiguous (1 KB); kt-tiles contiguous.
// Lane l holds column n = nt*16 + (l&15); its 16 values are
//   k = kt*32 + (l>>4)*8 + (e&7) + (e>=8 ? 16 : 0)     (mirrors A layout).
// ---------------------------------------------------------------------------
__global__ void pack_b_kernel(const float* __restrict__ X, bf16* __restrict__ out,
                              int K, int N, int ld, int trans, float scale) {
  int i = blockIdx.x * blockDim.x + threadIdx.x;
  int total = K * N;
  if (i >= total) return;
  int e    = i & 15;
  int lane = (i >> 4) & 31;
  int tile = i >> 9;
  int ktiles = K >> 5;
  int kt = tile % ktiles;
  int nt = tile / ktiles;
  int n = nt * 16 + (lane & 15);
  int k = kt * 32 + ((lane >> 4) << 3) + (e & 7) + ((e >= 8) ? 16 : 0);
  float v = trans ? X[(size_t)n * ld + k] : X[(size_t)k * ld + n];
  out[i] = (bf16)(v * scale);
}

// A fragment from row-major bf16 LDS (stride in elements).
__device__ inline v16bf load_a_lds_bf16(const bf16* base, int stride, int kt, int lane) {
  int row  = lane & 15;
  int koff = (lane >> 4) << 3;
  const bf16* p = base + row * stride + kt * 32 + koff;
  v8bf lo = *(const v8bf*)p;
  v8bf hi = *(const v8bf*)(p + 16);
  v16bf a;
#pragma unroll
  for (int i = 0; i < 8; i++) { a[i] = lo[i]; a[8 + i] = hi[i]; }
  return a;
}

// A fragment from row-major f32 LDS, converted to bf16.
__device__ inline v16bf load_a_lds_f32(const float* base, int stride, int kt, int lane) {
  int row  = lane & 15;
  int koff = (lane >> 4) << 3;
  const float* p = base + row * stride + kt * 32 + koff;
  v4f a0 = *(const v4f*)p;
  v4f a1 = *(const v4f*)(p + 4);
  v4f b0 = *(const v4f*)(p + 16);
  v4f b1 = *(const v4f*)(p + 20);
  v16bf a;
#pragma unroll
  for (int i = 0; i < 4; i++) {
    a[i]      = (bf16)a0[i];
    a[4 + i]  = (bf16)a1[i];
    a[8 + i]  = (bf16)b0[i];
    a[12 + i] = (bf16)b1[i];
  }
  return a;
}

// B fragment from the packed global array (L2/WGP$-resident stream).
__device__ inline v16bf load_b_tile(const bf16* __restrict__ packed, int ktiles,
                                    int nt, int kt, int lane) {
  const bf16* p = packed + (((size_t)nt * ktiles + kt) << 9) + (lane << 4);
  __builtin_prefetch(p + 512, 0, 3);  // next k-tile, WGP-scope prefetch
  return *(const v16bf*)p;
}

// NT independent accumulator tiles share each A fragment: 1 A-load (+cvt) per
// NT WMMAs, NT independent WMMA dependency chains for the matrix pipe.
template <int NT, int KT, bool AF32>
__device__ inline void gemm_group(const void* abase, int astride,
                                  const bf16* __restrict__ packed, int ktiles,
                                  int nt0, int lane, v8f (&acc)[NT]) {
#pragma unroll 8
  for (int kt = 0; kt < KT; kt++) {
    v16bf a = AF32 ? load_a_lds_f32((const float*)abase, astride, kt, lane)
                   : load_a_lds_bf16((const bf16*)abase, astride, kt, lane);
#pragma unroll
    for (int j = 0; j < NT; j++) {
      v16bf b = load_b_tile(packed, ktiles, nt0 + j, kt, lane);
      WMMA_BF16(acc[j], a, b);
    }
  }
}

// ---------------------------------------------------------------------------
// Main kernel. One workgroup = 8 waves = 256 threads owns 32 batch rows.
// LDS (dynamic, 168 KB):
//   Vt   f32 [32][512]   @ 0        (64 KB)  V iterate, resident all iters
//   Dt   f32 [32][512]   @ 64K      (64 KB)  momentum; aliases Hb in prologue
//   Tt   bf16[32][256]   @ 128K     (16 KB)  V @ S^T; aliases inb in prologue
//   Ut   bf16[32][128]   @ 144K     ( 8 KB)  relu(V Pin^T - Vin)
//   VinT f32 [32][128]   @ 152K     (16 KB)
// ---------------------------------------------------------------------------
__global__ __launch_bounds__(256)
void amn_qp_kernel(const float* __restrict__ input, const float* __restrict__ Vin,
                   const float* __restrict__ b1, const float* __restrict__ b2,
                   const bf16* __restrict__ W1p, const bf16* __restrict__ W2p,
                   const bf16* __restrict__ STp, const bf16* __restrict__ Sp,
                   const bf16* __restrict__ PinTp, const bf16* __restrict__ Pinp,
                   const int* __restrict__ n_iter_p, float* __restrict__ out) {
  extern __shared__ char smem[];
  float* Vt   = (float*)(smem);
  float* Dt   = (float*)(smem + 65536);
  bf16*  Hb   = (bf16*)(smem + 65536);    // alias of Dt (prologue only)
  bf16*  Tt   = (bf16*)(smem + 131072);
  bf16*  inb  = (bf16*)(smem + 131072);   // alias of Tt (prologue only)
  bf16*  Ut   = (bf16*)(smem + 147456);
  float* VinT = (float*)(smem + 155648);

  const int tid  = threadIdx.x;
  const int wave = tid >> 5;               // 0..7
  const int lane = tid & 31;
  const int row0 = blockIdx.x * 32;
  const int col  = lane & 15;
  const int rbase = (lane >> 4) << 3;
  const int mt   = wave >> 2;              // 0..1 : M-tile owned by this wave
  const int wq   = wave & 3;               // 0..3 : N-range selector

  // ---- load input tile (bf16) and Vin tile (f32) into LDS, coalesced ----
#pragma unroll 4
  for (int j = 0; j < 16; j++) {
    int e = j * 256 + tid;
    int r = e >> 7, c = e & 127;
    inb[e]  = (bf16)input[(size_t)(row0 + r) * 128 + c];
    VinT[e] = Vin[(size_t)(row0 + r) * 128 + c];
  }
  __syncthreads();

  // ---- hidden = relu(input @ W1 + b1) -> Hb bf16 [32][1024] ----
  // 2 mt x 64 nt tiles; this wave: mt, nt in [wq*16, wq*16+16), 4 at a time.
  for (int g = 0; g < 4; g++) {
    int nt0 = wq * 16 + g * 4;
    v8f acc[4];
#pragma unroll
    for (int j = 0; j < 4; j++) {
      float bv = b1[(nt0 + j) * 16 + col];
#pragma unroll
      for (int r = 0; r < 8; r++) acc[j][r] = bv;
    }
    gemm_group<4, 4, false>(inb, 128, W1p, 4, nt0, lane, acc);
#pragma unroll
    for (int j = 0; j < 4; j++) {
      int c0 = (nt0 + j) * 16 + col;
#pragma unroll
      for (int r = 0; r < 8; r++)
        Hb[(mt * 16 + rbase + r) * 1024 + c0] = (bf16)fmaxf(acc[j][r], 0.0f);
    }
  }
  __syncthreads();

  // ---- V0 = Hb @ W2 + b2 -> Vt f32 [32][512] ----
  // 2 mt x 32 nt; this wave: mt, nt in [wq*8, wq*8+8), 4 at a time.
  for (int g = 0; g < 2; g++) {
    int nt0 = wq * 8 + g * 4;
    v8f acc[4];
#pragma unroll
    for (int j = 0; j < 4; j++) {
      float bv = b2[(nt0 + j) * 16 + col];
#pragma unroll
      for (int r = 0; r < 8; r++) acc[j][r] = bv;
    }
    gemm_group<4, 32, false>(Hb + mt * 16 * 1024, 1024, W2p, 32, nt0, lane, acc);
#pragma unroll
    for (int j = 0; j < 4; j++) {
      int c0 = (nt0 + j) * 16 + col;
#pragma unroll
      for (int r = 0; r < 8; r++)
        Vt[(mt * 16 + rbase + r) * 512 + c0] = acc[j][r];
    }
  }
  __syncthreads();                      // Hb fully consumed; Dt may be zeroed

#pragma unroll 8
  for (int j = 0; j < 64; j++) Dt[j * 256 + tid] = 0.0f;

  const int n_iter = *n_iter_p;

  // ======================= momentum-GD iterations ==========================
  for (int it = 0; it < n_iter; it++) {
    __syncthreads();                    // Vt stable (prev update / prologue)

    // T = V @ S^T  [32x256] bf16 : 2 mt x 16 nt; wave: nt in [wq*4, wq*4+4)
    {
      int nt0 = wq * 4;
      v8f acc[4] = {};
      gemm_group<4, 16, true>(Vt + mt * 16 * 512, 512, STp, 16, nt0, lane, acc);
#pragma unroll
      for (int j = 0; j < 4; j++) {
        int c0 = (nt0 + j) * 16 + col;
#pragma unroll
        for (int r = 0; r < 8; r++)
          Tt[(mt * 16 + rbase + r) * 256 + c0] = (bf16)acc[j][r];
      }
    }

    // U = relu(V @ Pin^T - Vin) [32x128] bf16 : 2 mt x 8 nt; wave: 2 nt
    {
      int nt0 = wq * 2;
      v8f acc[2] = {};
      gemm_group<2, 16, true>(Vt + mt * 16 * 512, 512, PinTp, 16, nt0, lane, acc);
#pragma unroll
      for (int j = 0; j < 2; j++) {
        int c0 = (nt0 + j) * 16 + col;
#pragma unroll
        for (int r = 0; r < 8; r++) {
          int row = mt * 16 + rbase + r;
          float u = acc[j][r] - VinT[row * 128 + c0];
          Ut[row * 128 + c0] = (bf16)fmaxf(u, 0.0f);
        }
      }
    }
    __syncthreads();                    // Tt, Ut complete

    // dV = T @ (S*2/nm) + U @ (Pin*2/ni) + min(V,0)*2/nf ; momentum update
    // 2 mt x 32 nt; wave: nt in [wq*8, wq*8+8), 4 at a time.
    for (int g = 0; g < 2; g++) {
      int nt0 = wq * 8 + g * 4;
      v8f acc[4] = {};
      gemm_group<4, 8, false>(Tt + mt * 16 * 256, 256, Sp, 8, nt0, lane, acc);
      gemm_group<4, 4, false>(Ut + mt * 16 * 128, 128, Pinp, 4, nt0, lane, acc);
#pragma unroll
      for (int j = 0; j < 4; j++) {
        int c0 = (nt0 + j) * 16 + col;
#pragma unroll
        for (int r = 0; r < 8; r++) {
          int idx  = (mt * 16 + rbase + r) * 512 + c0;
          float v  = Vt[idx];
          float dv = acc[j][r] + fminf(v, 0.0f) * (2.0f / 512.0f);
          float d  = Dt[idx];
          d = 0.9f * d - 0.01f * dv;
          Dt[idx] = d;
          Vt[idx] = v + d;
        }
      }
    }
  }
  __syncthreads();

  // ---- write V tile to output, coalesced ----
#pragma unroll 4
  for (int j = 0; j < 64; j++) {
    int e = j * 256 + tid;
    int r = e >> 9, c = e & 511;
    out[(size_t)(row0 + r) * 512 + c] = Vt[e];
  }
}

// ---------------------------------------------------------------------------
extern "C" void kernel_launch(void* const* d_in, const int* in_sizes, int n_in,
                              void* d_out, int out_size, void* d_ws, size_t ws_size,
                              hipStream_t stream) {
  const float* input = (const float*)d_in[0];
  /* d_in[1] = Vref : unused by the reference math */
  const float* Vin = (const float*)d_in[2];
  const float* W1  = (const float*)d_in[3];
  const float* b1  = (const float*)d_in[4];
  const float* W2  = (const float*)d_in[5];
  const float* b2  = (const float*)d_in[6];
  const float* S   = (const float*)d_in[7];
  const float* Pin = (const float*)d_in[8];
  const int* n_it  = (const int*)d_in[9];
  float* out = (float*)d_out;

  // Packed bf16 B-operand arrays in workspace (2 MB total; L2-resident).
  bf16* W1p   = (bf16*)d_ws;           // 128x1024
  bf16* W2p   = W1p + 131072;          // 1024x512
  bf16* STp   = W2p + 524288;          // 512x256  (S^T)
  bf16* Sp    = STp + 131072;          // 256x512  (scaled 2/256)
  bf16* PinTp = Sp  + 131072;          // 512x128  (Pin^T)
  bf16* Pinp  = PinTp + 65536;         // 128x512  (scaled 2/128)

  auto pk = [&](const float* X, bf16* o, int K, int N, int ld, int tr, float sc) {
    int total = K * N;
    pack_b_kernel<<<(total + 255) / 256, 256, 0, stream>>>(X, o, K, N, ld, tr, sc);
  };
  pk(W1,  W1p,   128, 1024, 1024, 0, 1.0f);
  pk(W2,  W2p,  1024,  512,  512, 0, 1.0f);
  pk(S,   STp,   512,  256,  512, 1, 1.0f);
  pk(S,   Sp,    256,  512,  512, 0, 2.0f / 256.0f);
  pk(Pin, PinTp, 512,  128,  512, 1, 1.0f);
  pk(Pin, Pinp,  128,  512,  512, 0, 2.0f / 128.0f);

  const int lds_bytes = 172032;  // 168 KB of the WGP's 320 KB
  hipFuncSetAttribute((const void*)amn_qp_kernel,
                      hipFuncAttributeMaxDynamicSharedMemorySize, lds_bytes);

  amn_qp_kernel<<<32768 / 32, 256, lds_bytes, stream>>>(
      input, Vin, b1, b2, W1p, W2p, STp, Sp, PinTp, Pinp, n_it, out);
}
